// NCNC2_6545530159545
// MI455X (gfx1250) — matrix operand order repaired and checked by reading, not compiled, not run on hardware
//
#include <hip/hip_runtime.h>

// ---------------------------------------------------------------------------
// CDNA5 (gfx1250) fused link-prediction pipeline.
// GEMMs use v_wmma_f32_16x16x32_f16 (wave32), double-buffered K-loop so
// B-fragment global loads overlap WMMA issue. Hidden activations (16x512 per
// block) stay in LDS; LayerNorm stats via ds_add_f32 atomics.
// ---------------------------------------------------------------------------

typedef __attribute__((ext_vector_type(16))) _Float16 v16h;
typedef __attribute__((ext_vector_type(8)))  float    v8f;

union Frag {
    v16h  v;
    uint4 q[2];
    _Float16 h[16];
};

static constexpr int ND    = 128;                 // D
static constexpr int NFEAT = 256;                 // 2*D
static constexpr int HID   = 512;                 // hidden
static constexpr int BNUM  = 4096;                // batch of links
static constexpr int NWALK = 20;
static constexpr int MROWS = BNUM * NWALK;        // 81920
static constexpr int HPAD  = 520;                 // LDS row pad (banks)

// ---------------------------------------------------------------------------
// Weight convert + transpose: W[K,N] f32  ->  Wt[N,K] f16  (coalesced writes)
// ---------------------------------------------------------------------------
__global__ void k_convert_w(const float* __restrict__ W, _Float16* __restrict__ Wt,
                            int K, int N) {
    int idx = blockIdx.x * blockDim.x + threadIdx.x;
    if (idx >= K * N) return;
    int n = idx / K;
    int k = idx - n * K;
    Wt[idx] = (_Float16)W[k * N + n];
}

// pairs20[k,d] = emb[i[k],d] * emb[j[k],d]   (k < 20)
__global__ void k_pairs20(const int* __restrict__ tl, const float* __restrict__ emb,
                          float* __restrict__ pairs) {
    int idx = blockIdx.x * blockDim.x + threadIdx.x;
    if (idx >= NWALK * ND) return;
    int k = idx >> 7, d = idx & 127;
    int i = tl[k], j = tl[BNUM + k];
    pairs[idx] = emb[i * ND + d] * emb[j * ND + d];
}

// prod_ij[b,d] = emb[i[b],d] * emb[j[b],d]
__global__ void k_prodij(const int* __restrict__ tl, const float* __restrict__ emb,
                         float* __restrict__ prod) {
    int idx = blockIdx.x * blockDim.x + threadIdx.x;   // exactly B*128 threads
    int b = idx >> 7, d = idx & 127;
    int i = tl[b], j = tl[BNUM + b];
    prod[idx] = emb[i * ND + d] * emb[j * ND + d];
}

// z[row,d] = prod over 3 walk nodes; x[row,d] = f16(z * pairs20[row%20, d])
__global__ void k_gather_zx(const int* __restrict__ walks, const float* __restrict__ emb,
                            const float* __restrict__ pairs,
                            float* __restrict__ z, _Float16* __restrict__ x) {
    long idx = (long)blockIdx.x * blockDim.x + threadIdx.x;  // MROWS*128 threads
    long row = idx >> 7;
    int  d   = (int)(idx & 127);
    int  k   = (int)(row % NWALK);
    const int* w = walks + row * 3;
    float zz = emb[(long)w[0] * ND + d] * emb[(long)w[1] * ND + d] * emb[(long)w[2] * ND + d];
    z[idx] = zz;
    x[idx] = (_Float16)(zz * pairs[k * ND + d]);
}

// allcn[b,d] = sum_k P[b,k]*z[b,k,d];  final = f16 concat(prod_ij, allcn)
__global__ void k_allcn_final(const float* __restrict__ P, const float* __restrict__ z,
                              const float* __restrict__ prod, _Float16* __restrict__ fin) {
    int idx = blockIdx.x * blockDim.x + threadIdx.x;   // exactly B*128 threads
    int b = idx >> 7, d = idx & 127;
    long base = (long)b * NWALK;
    float s = 0.f;
    #pragma unroll
    for (int k = 0; k < NWALK; ++k)
        s += P[base + k] * z[(base + k) * ND + d];
    fin[b * NFEAT + d]      = (_Float16)prod[idx];
    fin[b * NFEAT + ND + d] = (_Float16)s;
}

// ---------------------------------------------------------------------------
// Double-buffered 16x128-per-wave GEMM inner loop.
//   A: LDS, rows [0,16) with row stride `astride` halves (fragment layout per
//      ISA 7.12.2: lane-half 0 holds K 0-7/16-23, lane-half 1 holds 8-15/24-31)
//   B: global Wt[N,K] f16, col n = nwv + t*16 + l16, same K split as A
// ---------------------------------------------------------------------------
template <int K>
__device__ __forceinline__ void gemm_dbuf(const _Float16* abase, int astride,
                                          const _Float16* __restrict__ wt,
                                          int l16, int hi, int nwv, v8f (&acc)[8])
{
    static_assert(K % 64 == 0, "K must be a multiple of 64");
    const _Float16* ap = abase + l16 * astride + hi * 8;
    const _Float16* bp = wt + (long)(nwv + l16) * K + hi * 8;

    Frag a0, a1, b0[8], b1[8];

    auto loadA = [&](Frag& a, int k0) {
        a.q[0] = *(const uint4*)(ap + k0);
        a.q[1] = *(const uint4*)(ap + k0 + 16);
    };
    auto loadB = [&](Frag (&b)[8], int k0) {
        #pragma unroll
        for (int t = 0; t < 8; ++t) {
            const _Float16* p = bp + (long)t * 16 * K + k0;
            b[t].q[0] = *(const uint4*)(p);
            b[t].q[1] = *(const uint4*)(p + 16);
        }
    };
    auto mma = [&](Frag& a, Frag (&b)[8]) {
        #pragma unroll
        for (int t = 0; t < 8; ++t)
            acc[t] = __builtin_amdgcn_wmma_f32_16x16x32_f16(
                false, a.v, false, b[t].v, (short)0, acc[t], false, false);
    };

    loadA(a0, 0);
    loadB(b0, 0);
    #pragma unroll
    for (int k0 = 0; k0 < K; k0 += 64) {
        if (k0 + 32 < K) { loadA(a1, k0 + 32); loadB(b1, k0 + 32); }
        mma(a0, b0);
        if (k0 + 64 < K) { loadA(a0, k0 + 64); loadB(b0, k0 + 64); }
        if (k0 + 32 < K) mma(a1, b1);
    }
}

// ---------------------------------------------------------------------------
// Fused 4-layer MLP:  X[M,KIN] -> relu+LN -> relu -> relu -> dot(W4)+b4
// One block = 16 rows. 4 waves; wave w owns output cols [128w, 128w+128).
// All intermediate 16x512 activations live in LDS (f16).
// ---------------------------------------------------------------------------
template <int KIN, bool SIG>
__global__ void __launch_bounds__(128) k_mlp_fused(
    const _Float16* __restrict__ X,
    const _Float16* __restrict__ W1t, const float* __restrict__ b1,
    const float*    __restrict__ g,   const float* __restrict__ bet,
    const _Float16* __restrict__ W2t, const float* __restrict__ b2,
    const _Float16* __restrict__ W3t, const float* __restrict__ b3,
    const float*    __restrict__ W4,  const float* __restrict__ b4,
    float* __restrict__ out)
{
    __shared__ _Float16 hb0[16][HPAD];
    __shared__ _Float16 hb1[16][HPAD];
    __shared__ float rsum[16], rsq[16], pv[16];

    const int tid  = threadIdx.x;
    const int wave = tid >> 5;
    const int lane = tid & 31;
    const int l16  = lane & 15;
    const int hi   = lane >> 4;       // 0: lanes 0-15, 1: lanes 16-31
    const int m0   = blockIdx.x * 16;
    const int nwv  = wave * 128;      // this wave's output-column base
    const int rb   = hi * 8;          // accumulator row base (C layout)

    // ---- stage the 16 x KIN A-tile into hb1 (coalesced b128, shared by waves)
    {
        const int chunks = 16 * KIN / 8;             // uint4 chunks
        for (int i = tid; i < chunks; i += 128) {
            int e   = i * 8;
            int row = e / KIN;
            int col = e - row * KIN;
            *(uint4*)&hb1[row][col] = *(const uint4*)(X + (long)(m0 + row) * KIN + col);
        }
    }
    if (tid < 16) { rsum[tid] = 0.f; rsq[tid] = 0.f; pv[tid] = 0.f; }
    __syncthreads();

    v8f acc[8];

    // ---------------- GEMM1: X @ W1  (+bias, ReLU, LayerNorm) -> hb0 --------
    #pragma unroll
    for (int t = 0; t < 8; ++t)
        #pragma unroll
        for (int e = 0; e < 8; ++e) acc[t][e] = 0.f;
    gemm_dbuf<KIN>(&hb1[0][0], HPAD, W1t, l16, hi, nwv, acc);

    #pragma unroll
    for (int v = 0; v < 8; ++v) {
        float s = 0.f, sq = 0.f;
        #pragma unroll
        for (int t = 0; t < 8; ++t) {
            const int col = nwv + t * 16 + l16;
            float val = acc[t][v] + b1[col];
            val = val > 0.f ? val : 0.f;
            acc[t][v] = val;
            s += val; sq += val * val;
        }
        atomicAdd(&rsum[rb + v], s);     // ds_add_f32
        atomicAdd(&rsq [rb + v], sq);
    }
    __syncthreads();
    #pragma unroll
    for (int v = 0; v < 8; ++v) {
        const int r = rb + v;
        const float mu   = rsum[r] * (1.f / 512.f);
        const float var  = rsq[r] * (1.f / 512.f) - mu * mu;
        const float rstd = rsqrtf(var + 1e-5f);
        #pragma unroll
        for (int t = 0; t < 8; ++t) {
            const int col = nwv + t * 16 + l16;
            hb0[r][col] = (_Float16)((acc[t][v] - mu) * rstd * g[col] + bet[col]);
        }
    }
    __syncthreads();   // hb0 ready; hb1 (staged X) no longer read after this

    // ---------------- GEMM2: hb0 @ W2 (+bias, ReLU) -> hb1 ----------------
    #pragma unroll
    for (int t = 0; t < 8; ++t)
        #pragma unroll
        for (int e = 0; e < 8; ++e) acc[t][e] = 0.f;
    gemm_dbuf<HID>(&hb0[0][0], HPAD, W2t, l16, hi, nwv, acc);
    #pragma unroll
    for (int v = 0; v < 8; ++v) {
        const int r = rb + v;
        #pragma unroll
        for (int t = 0; t < 8; ++t) {
            const int col = nwv + t * 16 + l16;
            float val = acc[t][v] + b2[col];
            hb1[r][col] = (_Float16)(val > 0.f ? val : 0.f);
        }
    }
    __syncthreads();

    // ---------------- GEMM3: hb1 @ W3 (+bias, ReLU) -> hb0 ----------------
    #pragma unroll
    for (int t = 0; t < 8; ++t)
        #pragma unroll
        for (int e = 0; e < 8; ++e) acc[t][e] = 0.f;
    gemm_dbuf<HID>(&hb1[0][0], HPAD, W3t, l16, hi, nwv, acc);
    #pragma unroll
    for (int v = 0; v < 8; ++v) {
        const int r = rb + v;
        #pragma unroll
        for (int t = 0; t < 8; ++t) {
            const int col = nwv + t * 16 + l16;
            float val = acc[t][v] + b3[col];
            hb0[r][col] = (_Float16)(val > 0.f ? val : 0.f);
        }
    }
    __syncthreads();

    // ---------------- W4 dot (+b4, optional sigmoid) ----------------
    {
        const int r   = tid >> 3;        // 16 rows x 8 threads
        const int seg = (tid & 7) * 64;  // 64 cols each
        float s = 0.f;
        #pragma unroll 8
        for (int c = seg; c < seg + 64; ++c)
            s += (float)hb0[r][c] * W4[c];
        atomicAdd(&pv[r], s);
    }
    __syncthreads();
    if (tid < 16) {
        float val = pv[tid] + b4[0];
        if (SIG) val = 1.f / (1.f + __expf(-val));
        out[m0 + tid] = val;
    }
}

// ---------------------------------------------------------------------------
extern "C" void kernel_launch(void* const* d_in, const int* in_sizes, int n_in,
                              void* d_out, int out_size, void* d_ws, size_t ws_size,
                              hipStream_t stream) {
    (void)in_sizes; (void)n_in; (void)out_size; (void)ws_size;

    const int*   tl    = (const int*)d_in[0];
    const int*   walks = (const int*)d_in[1];
    const float* emb   = (const float*)d_in[2];
    // ncn params
    const float* nW1 = (const float*)d_in[3];
    const float* nb1 = (const float*)d_in[4];
    const float* ngm = (const float*)d_in[5];
    const float* nbe = (const float*)d_in[6];
    const float* nW2 = (const float*)d_in[7];
    const float* nb2 = (const float*)d_in[8];
    const float* nW3 = (const float*)d_in[9];
    const float* nb3 = (const float*)d_in[10];
    const float* nW4 = (const float*)d_in[11];
    const float* nb4 = (const float*)d_in[12];
    // mlp params
    const float* mW1 = (const float*)d_in[13];
    const float* mb1 = (const float*)d_in[14];
    const float* mgm = (const float*)d_in[15];
    const float* mbe = (const float*)d_in[16];
    const float* mW2 = (const float*)d_in[17];
    const float* mb2 = (const float*)d_in[18];
    const float* mW3 = (const float*)d_in[19];
    const float* mb3 = (const float*)d_in[20];
    const float* mW4 = (const float*)d_in[21];
    const float* mb4 = (const float*)d_in[22];

    // ---- workspace layout (256B aligned) ----
    char* ws = (char*)d_ws;
    size_t off = 0;
    auto carve = [&](size_t bytes) -> char* {
        char* p = ws + off;
        off = (off + bytes + 255) & ~(size_t)255;
        return p;
    };
    _Float16* ncnW1t = (_Float16*)carve((size_t)ND    * HID * 2);
    _Float16* ncnW2t = (_Float16*)carve((size_t)HID   * HID * 2);
    _Float16* ncnW3t = (_Float16*)carve((size_t)HID   * HID * 2);
    _Float16* mlpW1t = (_Float16*)carve((size_t)NFEAT * HID * 2);
    _Float16* mlpW2t = (_Float16*)carve((size_t)HID   * HID * 2);
    _Float16* mlpW3t = (_Float16*)carve((size_t)HID   * HID * 2);
    float*    pairs  = (float*)   carve((size_t)NWALK * ND  * 4);
    float*    prod   = (float*)   carve((size_t)BNUM  * ND  * 4);
    float*    zbuf   = (float*)   carve((size_t)MROWS * ND  * 4);
    _Float16* xbuf   = (_Float16*)carve((size_t)MROWS * ND  * 2);
    float*    Pbuf   = (float*)   carve((size_t)MROWS * 4);
    _Float16* finbuf = (_Float16*)carve((size_t)BNUM  * NFEAT * 2);

    // ---- 1) weight conversion (f32 [K,N] -> f16 [N,K]) ----
    k_convert_w<<<(ND * HID + 255) / 256,    256, 0, stream>>>(nW1, ncnW1t, ND,    HID);
    k_convert_w<<<(HID * HID + 255) / 256,   256, 0, stream>>>(nW2, ncnW2t, HID,   HID);
    k_convert_w<<<(HID * HID + 255) / 256,   256, 0, stream>>>(nW3, ncnW3t, HID,   HID);
    k_convert_w<<<(NFEAT * HID + 255) / 256, 256, 0, stream>>>(mW1, mlpW1t, NFEAT, HID);
    k_convert_w<<<(HID * HID + 255) / 256,   256, 0, stream>>>(mW2, mlpW2t, HID,   HID);
    k_convert_w<<<(HID * HID + 255) / 256,   256, 0, stream>>>(mW3, mlpW3t, HID,   HID);

    // ---- 2) gathers / elementwise ----
    k_pairs20<<<(NWALK * ND + 255) / 256, 256, 0, stream>>>(tl, emb, pairs);
    k_prodij <<<(BNUM * ND) / 256,        256, 0, stream>>>(tl, emb, prod);
    k_gather_zx<<<((long)MROWS * ND) / 256, 256, 0, stream>>>(walks, emb, pairs, zbuf, xbuf);

    // ---- 3) ncn MLP (fused, WMMA) -> P = sigmoid(...) on 81920 rows ----
    k_mlp_fused<ND, true><<<MROWS / 16, 128, 0, stream>>>(
        xbuf, ncnW1t, nb1, ngm, nbe, ncnW2t, nb2, ncnW3t, nb3, nW4, nb4, Pbuf);

    // ---- 4) allcn + concat -> final [B, 256] f16 ----
    k_allcn_final<<<(BNUM * ND) / 256, 256, 0, stream>>>(Pbuf, zbuf, prod, finbuf);

    // ---- 5) final MLP (fused, WMMA) -> d_out [B,1] f32 ----
    k_mlp_fused<NFEAT, false><<<BNUM / 16, 128, 0, stream>>>(
        finbuf, mlpW1t, mb1, mgm, mbe, mlpW2t, mb2, mlpW3t, mb3, mW4, mb4, (float*)d_out);
}
